// MambaBlock_72146860638277
// MI455X (gfx1250) — matrix operand order repaired and checked by reading, not compiled, not run on hardware
//
#include <hip/hip_runtime.h>
#include <cstdint>

#define D_MODEL 1024
#define D_STATE 16
#define D_CONV  4
#define D_INNER 2048
#define BB      2
#define LL      2048
#define DT_C    0.1f
#define NCHUNK  16
#define TCHUNK  (LL / NCHUNK)   /* 128 */

typedef __attribute__((ext_vector_type(16))) __bf16 v16bf;
typedef __attribute__((ext_vector_type(8)))  float  v8f;

// ---- fp32 -> bf16 (RNE) helpers for split-bf16 precomputation ----
__device__ __forceinline__ unsigned bf_round_bits(float x) {
  unsigned u = __float_as_uint(x);
  u += 0x7FFFu + ((u >> 16) & 1u);
  return u;
}
__device__ __forceinline__ unsigned short bfbits(float x) {
  return (unsigned short)(bf_round_bits(x) >> 16);
}
__device__ __forceinline__ float bf_hi_f(float x) {
  return __uint_as_float(bf_round_bits(x) & 0xFFFF0000u);
}

// =====================================================================
// Elementwise split: src(f32) -> hi(bf16 bits), lo(bf16 bits of residual)
// 4 elements per thread.
// =====================================================================
__global__ __launch_bounds__(256)
void split_bf16_kernel(const float* __restrict__ src,
                       unsigned short* __restrict__ hi,
                       unsigned short* __restrict__ lo)
{
  int i = blockIdx.x * 256 + threadIdx.x;     // index in float4 units
  float4 v = ((const float4*)src)[i];
  uint2 hv, lv;
  hv.x = (unsigned)bfbits(v.x) | ((unsigned)bfbits(v.y) << 16);
  hv.y = (unsigned)bfbits(v.z) | ((unsigned)bfbits(v.w) << 16);
  lv.x = (unsigned)bfbits(v.x - bf_hi_f(v.x)) | ((unsigned)bfbits(v.y - bf_hi_f(v.y)) << 16);
  lv.y = (unsigned)bfbits(v.z - bf_hi_f(v.z)) | ((unsigned)bfbits(v.w - bf_hi_f(v.w)) << 16);
  ((uint2*)hi)[i] = hv;
  ((uint2*)lo)[i] = lv;
}

// =====================================================================
// C[M,N] = A[M,K] @ W[N,K]^T, operands pre-split into bf16 hi/lo planes.
// One wave -> 32x64 output tile (2 A fragments x 4 B tiles, 8 accums).
// Inner loop per k-step: 24 b128 loads, 24 x v_wmma_f32_16x16x32_bf16.
// =====================================================================
union BV { v16bf v; float4 f[2]; };

__global__ __launch_bounds__(256)
void gemm_bf16x2_wmma(const unsigned short* __restrict__ Ahi,
                      const unsigned short* __restrict__ Alo,
                      const unsigned short* __restrict__ Whi,
                      const unsigned short* __restrict__ Wlo,
                      float* __restrict__ C, int M, int N, int K)
{
  const int lane = threadIdx.x;        // 0..31
  const int wave = threadIdx.y;        // 0..7
  const int half = lane >> 4;
  const int mr   = lane & 15;
  const int m0   = (blockIdx.x * 8 + wave) * 32;
  const int nb   = blockIdx.y * 64;
  if (m0 >= M) return;

  const int arow0 = m0 + mr;
  const int arow1 = m0 + 16 + mr;
  int wrow[4];
#pragma unroll
  for (int t = 0; t < 4; ++t) {
    int r = nb + t * 16 + mr;
    wrow[t] = (r < N) ? r : (N - 1);   // clamp keeps loads in-bounds, EXEC uniform
  }

  v8f acc0[4] = {{}, {}, {}, {}};
  v8f acc1[4] = {{}, {}, {}, {}};

  for (int k0 = 0; k0 < K; k0 += 32) {
    // A fragments (16x32 bf16 layout): elems 0..7 at k0+half*8, 8..15 at +16
    size_t ab0 = (size_t)arow0 * K + k0 + half * 8;
    size_t ab1 = (size_t)arow1 * K + k0 + half * 8;
    BV ahi0, alo0, ahi1, alo1;
    ahi0.f[0] = *(const float4*)(Ahi + ab0);
    ahi0.f[1] = *(const float4*)(Ahi + ab0 + 16);
    alo0.f[0] = *(const float4*)(Alo + ab0);
    alo0.f[1] = *(const float4*)(Alo + ab0 + 16);
    ahi1.f[0] = *(const float4*)(Ahi + ab1);
    ahi1.f[1] = *(const float4*)(Ahi + ab1 + 16);
    alo1.f[0] = *(const float4*)(Alo + ab1);
    alo1.f[1] = *(const float4*)(Alo + ab1 + 16);

#pragma unroll
    for (int t = 0; t < 4; ++t) {
      // B fragment (32x16 bf16 layout): elems 0..7 at k0+half*8, 8..15 at +16
      size_t wbase = (size_t)wrow[t] * K + k0 + half * 8;
      BV bhi, blo;
      bhi.f[0] = *(const float4*)(Whi + wbase);
      bhi.f[1] = *(const float4*)(Whi + wbase + 16);
      blo.f[0] = *(const float4*)(Wlo + wbase);
      blo.f[1] = *(const float4*)(Wlo + wbase + 16);
      // split products: hi*hi + lo*hi + hi*lo  (~fp32 fidelity, f32 accumulate)
      acc0[t] = __builtin_amdgcn_wmma_f32_16x16x32_bf16(false, ahi0.v, false, bhi.v, (short)0, acc0[t], false, false);
      acc0[t] = __builtin_amdgcn_wmma_f32_16x16x32_bf16(false, alo0.v, false, bhi.v, (short)0, acc0[t], false, false);
      acc0[t] = __builtin_amdgcn_wmma_f32_16x16x32_bf16(false, ahi0.v, false, blo.v, (short)0, acc0[t], false, false);
      acc1[t] = __builtin_amdgcn_wmma_f32_16x16x32_bf16(false, ahi1.v, false, bhi.v, (short)0, acc1[t], false, false);
      acc1[t] = __builtin_amdgcn_wmma_f32_16x16x32_bf16(false, alo1.v, false, bhi.v, (short)0, acc1[t], false, false);
      acc1[t] = __builtin_amdgcn_wmma_f32_16x16x32_bf16(false, ahi1.v, false, blo.v, (short)0, acc1[t], false, false);
    }
  }

  // C/D layout: reg r -> M = base + half*8 + r, N = n0 + (lane&15)
#pragma unroll
  for (int t = 0; t < 4; ++t) {
    int ncol = nb + t * 16 + mr;
    if (ncol < N) {
#pragma unroll
      for (int r = 0; r < 8; ++r) {
        C[(size_t)(m0 + half * 8 + r) * N + ncol]      = acc0[t][r];
        C[(size_t)(m0 + 16 + half * 8 + r) * N + ncol] = acc1[t][r];
      }
    }
  }
}

// =====================================================================
// Depthwise causal conv1d (k=4, left pad 3) + bias + SiLU.
// Emits xs in f32 (for the scan) and as split-bf16 planes (for x_proj).
// =====================================================================
__global__ __launch_bounds__(256)
void conv_silu_kernel(const float* __restrict__ xz, const float* __restrict__ cw,
                      const float* __restrict__ cb, float* __restrict__ xs_act,
                      unsigned short* __restrict__ xs_hi, unsigned short* __restrict__ xs_lo)
{
  long idx = (long)blockIdx.x * 256 + threadIdx.x;       // total = 2^23
  int d = (int)(idx & (D_INNER - 1));
  int l = (int)((idx >> 11) & (LL - 1));
  int b = (int)(idx >> 22);
  float acc = cb[d];
#pragma unroll
  for (int j = 0; j < D_CONV; ++j) {
    int ls = l - (D_CONV - 1) + j;
    if (ls >= 0)
      acc += cw[d * D_CONV + j] * xz[(size_t)(b * LL + ls) * (2 * D_INNER) + d];
  }
  float s = acc / (1.0f + expf(-acc));
  size_t o = (size_t)(b * LL + l) * D_INNER + d;
  xs_act[o] = s;
  xs_hi[o] = bfbits(s);
  xs_lo[o] = bfbits(s - bf_hi_f(s));
}

// =====================================================================
// Chunked linear scan (dA time-invariant since dt is constant).
// =====================================================================
__global__ __launch_bounds__(256)
void scan_passA(const float* __restrict__ xs_act, const float* __restrict__ xdbl,
                const float* __restrict__ A_log, float* __restrict__ hend)
{
  int tid = blockIdx.x * 256 + threadIdx.x;   // 65536 = B*NC*D_INNER
  int d = tid & (D_INNER - 1);
  int c = (tid >> 11) & (NCHUNK - 1);
  int b = tid >> 15;
  float dA[D_STATE], h[D_STATE];
#pragma unroll
  for (int n = 0; n < D_STATE; ++n) {
    dA[n] = expf(-DT_C * expf(A_log[d * D_STATE + n]));
    h[n] = 0.0f;
  }
  int l0 = c * TCHUNK;
  for (int l = l0; l < l0 + TCHUNK; ++l) {
    size_t row = (size_t)(b * LL + l);
    float x = xs_act[row * D_INNER + d];
    float dtx = DT_C * x;
    const float* Bv = xdbl + row * (2 * D_STATE);
#pragma unroll
    for (int n = 0; n < D_STATE; ++n)
      h[n] = fmaf(dA[n], h[n], Bv[n] * dtx);
  }
  size_t base = ((size_t)(b * NCHUNK + c) * D_INNER + d) * D_STATE;
#pragma unroll
  for (int n = 0; n < D_STATE; ++n) hend[base + n] = h[n];
}

// Pass B: sequential combine over chunks: H_c = dA^T * H_{c-1} + hend_c
__global__ __launch_bounds__(256)
void scan_passB(const float* __restrict__ A_log, const float* __restrict__ hend,
                float* __restrict__ carryin)
{
  int tid = blockIdx.x * 256 + threadIdx.x;   // 4096 = B*D_INNER
  int d = tid & (D_INNER - 1);
  int b = tid >> 11;
  float dAT[D_STATE], carry[D_STATE];
#pragma unroll
  for (int n = 0; n < D_STATE; ++n) {
    float dA = expf(-DT_C * expf(A_log[d * D_STATE + n]));
    float p = dA;
#pragma unroll
    for (int i = 0; i < 7; ++i) p *= p;       // dA^128 = dA^TCHUNK
    dAT[n] = p;
    carry[n] = 0.0f;
  }
  for (int c = 0; c < NCHUNK; ++c) {
    size_t base = ((size_t)(b * NCHUNK + c) * D_INNER + d) * D_STATE;
#pragma unroll
    for (int n = 0; n < D_STATE; ++n) {
      carryin[base + n] = carry[n];
      carry[n] = fmaf(dAT[n], carry[n], hend[base + n]);
    }
  }
}

// Pass C: re-scan with carry-in; fuse y = h.C + x*D then * silu(z);
// emit y directly as split-bf16 planes (consumed only by out_proj GEMM).
__global__ __launch_bounds__(256)
void scan_passC(const float* __restrict__ xs_act, const float* __restrict__ xdbl,
                const float* __restrict__ xz, const float* __restrict__ A_log,
                const float* __restrict__ Dp, const float* __restrict__ carryin,
                unsigned short* __restrict__ y_hi, unsigned short* __restrict__ y_lo)
{
  int tid = blockIdx.x * 256 + threadIdx.x;
  int d = tid & (D_INNER - 1);
  int c = (tid >> 11) & (NCHUNK - 1);
  int b = tid >> 15;
  float dA[D_STATE], h[D_STATE];
  size_t cbase = ((size_t)(b * NCHUNK + c) * D_INNER + d) * D_STATE;
#pragma unroll
  for (int n = 0; n < D_STATE; ++n) {
    dA[n] = expf(-DT_C * expf(A_log[d * D_STATE + n]));
    h[n] = carryin[cbase + n];
  }
  float Dd = Dp[d];
  int l0 = c * TCHUNK;
  for (int l = l0; l < l0 + TCHUNK; ++l) {
    size_t row = (size_t)(b * LL + l);
    float x = xs_act[row * D_INNER + d];
    float dtx = DT_C * x;
    const float* Bv = xdbl + row * (2 * D_STATE);
    const float* Cv = Bv + D_STATE;
    float y = 0.0f;
#pragma unroll
    for (int n = 0; n < D_STATE; ++n) {
      h[n] = fmaf(dA[n], h[n], Bv[n] * dtx);
      y = fmaf(h[n], Cv[n], y);
    }
    y = fmaf(x, Dd, y);
    float z = xz[row * (2 * D_INNER) + D_INNER + d];
    y *= z / (1.0f + expf(-z));
    size_t o = row * D_INNER + d;
    y_hi[o] = bfbits(y);
    y_lo[o] = bfbits(y - bf_hi_f(y));
  }
}

// =====================================================================
extern "C" void kernel_launch(void* const* d_in, const int* in_sizes, int n_in,
                              void* d_out, int out_size, void* d_ws, size_t ws_size,
                              hipStream_t stream)
{
  const float* x     = (const float*)d_in[0];   // (B,L,1024)
  const float* Wi    = (const float*)d_in[1];   // (4096,1024)
  const float* cw    = (const float*)d_in[2];   // (2048,1,4)
  const float* cb    = (const float*)d_in[3];   // (2048,)
  const float* Wx    = (const float*)d_in[4];   // (32,2048)
  const float* A_log = (const float*)d_in[5];   // (2048,16)
  const float* Dp    = (const float*)d_in[6];   // (2048,)
  const float* Wo    = (const float*)d_in[7];   // (1024,2048)
  float* out = (float*)d_out;

  const size_t M = (size_t)BB * LL;             // 4096 rows
  char* p = (char*)d_ws;
  // f32 scratch
  float* xz      = (float*)p;                      p += M * (2 * D_INNER) * 4;        // 64 MB
  float* xs_act  = (float*)p;                      p += M * D_INNER * 4;              // 32 MB
  float* xdbl    = (float*)p;                      p += M * (2 * D_STATE) * 4;        // 0.5 MB
  float* hend    = (float*)p;                      p += (size_t)BB * NCHUNK * D_INNER * D_STATE * 4;
  float* carryin = (float*)p;                      p += (size_t)BB * NCHUNK * D_INNER * D_STATE * 4;
  // split-bf16 planes (bit patterns)
  unsigned short* xhi  = (unsigned short*)p;       p += M * D_MODEL * 2;
  unsigned short* xlo  = (unsigned short*)p;       p += M * D_MODEL * 2;
  unsigned short* wihi = (unsigned short*)p;       p += (size_t)(2 * D_INNER) * D_MODEL * 2;
  unsigned short* wilo = (unsigned short*)p;       p += (size_t)(2 * D_INNER) * D_MODEL * 2;
  unsigned short* xshi = (unsigned short*)p;       p += M * D_INNER * 2;   // reused as y_hi
  unsigned short* xslo = (unsigned short*)p;       p += M * D_INNER * 2;   // reused as y_lo
  unsigned short* wxhi = (unsigned short*)p;       p += (size_t)(2 * D_STATE) * D_INNER * 2;
  unsigned short* wxlo = (unsigned short*)p;       p += (size_t)(2 * D_STATE) * D_INNER * 2;
  unsigned short* wohi = (unsigned short*)p;       p += (size_t)D_MODEL * D_INNER * 2;
  unsigned short* wolo = (unsigned short*)p;       p += (size_t)D_MODEL * D_INNER * 2;

  dim3 blk(32, 8, 1);

  // 0) one-shot split-bf16 conversions (memory-bound elementwise)
  split_bf16_kernel<<<(unsigned)(M * D_MODEL / 4 / 256), 256, 0, stream>>>(x, xhi, xlo);
  split_bf16_kernel<<<(unsigned)((size_t)2 * D_INNER * D_MODEL / 4 / 256), 256, 0, stream>>>(Wi, wihi, wilo);
  split_bf16_kernel<<<(unsigned)((size_t)2 * D_STATE * D_INNER / 4 / 256), 256, 0, stream>>>(Wx, wxhi, wxlo);
  split_bf16_kernel<<<(unsigned)((size_t)D_MODEL * D_INNER / 4 / 256), 256, 0, stream>>>(Wo, wohi, wolo);

  // 1) in_proj: xz = x @ Wi^T   (M=4096, N=4096, K=1024); 256 rows/block
  gemm_bf16x2_wmma<<<dim3((unsigned)(M / 256), (2 * D_INNER) / 64), blk, 0, stream>>>(
      xhi, xlo, wihi, wilo, xz, (int)M, 2 * D_INNER, D_MODEL);

  // 2) depthwise causal conv + SiLU (also emits xs split planes)
  conv_silu_kernel<<<(BB * LL * D_INNER) / 256, 256, 0, stream>>>(xz, cw, cb, xs_act, xshi, xslo);

  // 3) x_proj: xdbl = xs @ Wx^T   (N=32 < 64-strip -> clamped/masked)
  gemm_bf16x2_wmma<<<dim3((unsigned)(M / 256), 1), blk, 0, stream>>>(
      xshi, xslo, wxhi, wxlo, xdbl, (int)M, 2 * D_STATE, D_INNER);

  // 4) chunked linear SSM scan
  scan_passA<<<(BB * NCHUNK * D_INNER) / 256, 256, 0, stream>>>(xs_act, xdbl, A_log, hend);
  scan_passB<<<(BB * D_INNER) / 256, 256, 0, stream>>>(A_log, hend, carryin);
  scan_passC<<<(BB * NCHUNK * D_INNER) / 256, 256, 0, stream>>>(
      xs_act, xdbl, xz, A_log, Dp, carryin, xshi, xslo);   // y splits overwrite xs splits

  // 5) out_proj: out = y @ Wo^T   (M=4096, N=1024, K=2048)
  gemm_bf16x2_wmma<<<dim3((unsigned)(M / 256), D_MODEL / 64), blk, 0, stream>>>(
      xshi, xslo, wohi, wolo, out, (int)M, D_MODEL, D_INNER);

  (void)in_sizes; (void)n_in; (void)out_size; (void)ws_size;
}
